// MultiHeadAttention_63041529971099
// MI455X (gfx1250) — matrix operand order repaired and checked by reading, not compiled, not run on hardware
//
#include <hip/hip_runtime.h>

// ---------------- problem constants ----------------
constexpr int Bc = 2, Sc = 2048, Dc = 768, Hc = 12, DKc = 64;

typedef __bf16 bf16_t;
typedef __attribute__((ext_vector_type(16))) __bf16 v16bf;
typedef __attribute__((ext_vector_type(8)))  __bf16 v8bf;
typedef __attribute__((ext_vector_type(8)))  float  v8f;

#define DEV static __device__ __forceinline__

// Native conversion: lowers to v_cvt_*bf16_f32 on gfx1250 (RNE).
DEV bf16_t f2bf(float f) { return (bf16_t)f; }

// A-operand (16x32 bf16), own k order: lane holds row m = lane&15,
// elements 0..15 -> k0 + half*16 + i (contiguous, matches loadBT)
DEV v16bf loadA(const bf16_t* __restrict__ base, int ld, int m, int k0, int half) {
  v8bf lo = *(const v8bf*)(base + (size_t)m * ld + k0 + half * 16);
  v8bf hi = *(const v8bf*)(base + (size_t)m * ld + k0 + half * 16 + 8);
  v16bf r;
#pragma unroll
  for (int i = 0; i < 8; i++) { r[i] = lo[i]; r[i + 8] = hi[i]; }
  return r;
}

// B-operand (32x16 bf16): lane holds col n = lane&15, k = half*16 + 0..15
// contiguous; caller passes pointer to the 16 contiguous elements.
DEV v16bf loadBT(const bf16_t* __restrict__ p) {
  v8bf lo = *(const v8bf*)p;
  v8bf hi = *(const v8bf*)(p + 8);
  v16bf r;
#pragma unroll
  for (int i = 0; i < 8; i++) { r[i] = lo[i]; r[i + 8] = hi[i]; }
  return r;
}

DEV v8f wmma_bf16(v16bf a, v16bf b, v8f c) {
  return __builtin_amdgcn_wmma_f32_16x16x32_bf16(false, a, false, b, (short)0, c,
                                                 false, false);
}

// ---------------- kernel 0a: bulk f32 -> bf16 convert of q,k,v ----------------
// xb: [3][B][S][768] bf16, same layout as inputs. 8 elements per thread.
__global__ __launch_bounds__(256) void mha_cvt(const float* __restrict__ q,
                                               const float* __restrict__ k,
                                               const float* __restrict__ v,
                                               bf16_t* __restrict__ xb) {
  const int BSD = Bc * Sc * Dc;
  int i8 = (blockIdx.x * blockDim.x + threadIdx.x) * 8;
  if (i8 >= 3 * BSD) return;
  int t = i8 / BSD;
  int off = i8 - t * BSD;
  const float* src = ((t == 0) ? q : (t == 1) ? k : v) + off;
  v8bf o;
#pragma unroll
  for (int i = 0; i < 8; i++) o[i] = f2bf(src[i]);
  *(v8bf*)(xb + i8) = o;
}

// ---------------- kernel 0b: weight prep (f32 -> bf16, transpose) ----------------
// wt : [3][H][64(n)][768(k)]  bf16 (B-operand layout for projections)
// wot: [768(n)][768(k)]       bf16 (B-operand layout for output proj)
__global__ void mha_prep(const float* __restrict__ Wq, const float* __restrict__ Wk,
                         const float* __restrict__ Wv, const float* __restrict__ Wo,
                         bf16_t* __restrict__ wt, bf16_t* __restrict__ wot) {
  const int NWT = 3 * Hc * DKc * Dc;
  const int NWO = Dc * Dc;
  for (int i = blockIdx.x * blockDim.x + threadIdx.x; i < NWT + NWO;
       i += gridDim.x * blockDim.x) {
    if (i < NWT) {
      int k = i % Dc; int t = i / Dc;
      int n = t % DKc; t /= DKc;
      int h = t % Hc; int p = t / Hc;
      const float* Ws = (p == 0) ? Wq : (p == 1) ? Wk : Wv;
      wt[i] = f2bf(Ws[((size_t)h * Dc + k) * DKc + n]);
    } else {
      int j = i - NWT;              // j = n*768 + k
      int k = j % Dc; int n = j / Dc;
      wot[j] = f2bf(Wo[(size_t)k * Dc + n]);
    }
  }
}

// ---------------- kernel 1: fused Q/K/V projections ----------------
// grid: ((p*H + h)*B + b)*(S/16) + sb  , block = 128 (4 waves, wave = 16 dk cols)
// Q projection output is pre-scaled by 1/sqrt(DK) = 0.125.
__global__ __launch_bounds__(128) void mha_proj(
    const bf16_t* __restrict__ xb, const float* __restrict__ bq,
    const float* __restrict__ bk, const float* __restrict__ bv,
    const bf16_t* __restrict__ wt, bf16_t* __restrict__ qh,
    bf16_t* __restrict__ kh, bf16_t* __restrict__ vt) {
  const int lane = threadIdx.x & 31;
  const int w = threadIdx.x >> 5;
  const int half = lane >> 4;
  const int nl = lane & 15;

  int idx = blockIdx.x;
  const int NSB = Sc / 16;
  int sb = idx % NSB; idx /= NSB;
  int b = idx % Bc; idx /= Bc;
  int h = idx % Hc;
  int p = idx / Hc;

  const float* bias = (p == 0) ? bq : (p == 1) ? bk : bv;
  const bf16_t* X = xb + ((size_t)p * Bc + b) * Sc * Dc;
  const bf16_t* wb = wt + ((size_t)(p * Hc + h)) * DKc * Dc;

  const int s0 = sb * 16;
  const int m = s0 + nl;
  const int ncol = w * 16 + nl;                       // dk column 0..63
  const bf16_t* brow = wb + (size_t)ncol * Dc;

  v8f acc = {};
  for (int kc = 0; kc < Dc; kc += 32) {
    v16bf a = loadA(X, Dc, m, kc, half);
    v16bf bm = loadBT(brow + kc + half * 16);
    acc = wmma_bf16(a, bm, acc);
  }
  const float bi = bias[h * DKc + ncol];
  const float osc = (p == 0) ? 0.125f : 1.0f;         // fold score scale into Q
  const size_t bh = (size_t)b * Hc + h;

  if (p < 2) {                                        // row-major [bh][s][64]
    bf16_t* dst = ((p == 0) ? qh : kh) + bh * Sc * DKc;
#pragma unroll
    for (int r = 0; r < 8; r++) {
      int row = s0 + r + 8 * half;
      dst[(size_t)row * DKc + ncol] = f2bf((acc[r] + bi) * osc);
    }
  } else {                                            // transposed [bh][dk][S]
    bf16_t* dst = vt + (bh * DKc + ncol) * Sc;
    v8bf pk;
#pragma unroll
    for (int r = 0; r < 8; r++) pk[r] = f2bf(acc[r] + bi);
    *(v8bf*)(dst + s0 + 8 * half) = pk;
  }
}

// ---------------- kernel 2: wave-level flash attention ----------------
// grid = B*H*(S/64), block = 128; each wave owns a 16-row query block
__global__ __launch_bounds__(128) void mha_attn(
    const bf16_t* __restrict__ qh, const bf16_t* __restrict__ kh,
    const bf16_t* __restrict__ vt, const int* __restrict__ maskp,
    bf16_t* __restrict__ obuf) {
  __shared__ __align__(16) bf16_t plds[4][16 * 32];
  const int lane = threadIdx.x & 31;
  const int w = threadIdx.x >> 5;
  const int half = lane >> 4;
  const int nl = lane & 15;

  int idx = blockIdx.x;
  const int NQB = Sc / 64;
  int t = idx % NQB;
  int bh = idx / NQB;
  int b = bh / Hc;
  int h = bh % Hc;
  int s0 = (t * 4 + w) * 16;

  const bf16_t* Q = qh + (size_t)bh * Sc * DKc;
  const bf16_t* K = kh + (size_t)bh * Sc * DKc;
  const bf16_t* V = vt + (size_t)bh * DKc * Sc;
  const int* mk = maskp + (size_t)b * Sc;

  v16bf aq0 = loadA(Q, DKc, s0 + nl, 0, half);
  v16bf aq1 = loadA(Q, DKc, s0 + nl, 32, half);

  v8f acc0 = {}, acc1 = {}, acc2 = {}, acc3 = {};
  float mrow[8], lrow[8];
#pragma unroll
  for (int r = 0; r < 8; r++) { mrow[r] = -3.0e38f; lrow[r] = 0.f; }

  bf16_t* pb = &plds[w][0];

  for (int kt = 0; kt < Sc; kt += 32) {
    // ---- scores: 16x32 logits via 4 WMMAs (scale pre-folded into Q) ----
    v8f sc0 = {}, sc1 = {};
    {
      const bf16_t* kp0 = K + (size_t)(kt + nl) * DKc;
      sc0 = wmma_bf16(aq0, loadBT(kp0 + half * 16), sc0);
      sc0 = wmma_bf16(aq1, loadBT(kp0 + 32 + half * 16), sc0);
      const bf16_t* kp1 = K + (size_t)(kt + 16 + nl) * DKc;
      sc1 = wmma_bf16(aq0, loadBT(kp1 + half * 16), sc1);
      sc1 = wmma_bf16(aq1, loadBT(kp1 + 32 + half * 16), sc1);
    }
    const bool mv0 = mk[kt + nl] != 0;
    const bool mv1 = mk[kt + 16 + nl] != 0;
    float s0a[8], s1a[8];
#pragma unroll
    for (int r = 0; r < 8; r++) {
      s0a[r] = mv0 ? sc0[r] : -3.0e38f;
      s1a[r] = mv1 ? sc1[r] : -3.0e38f;
    }
    // ---- online softmax: row reductions across 16-lane halves ----
    float tm[8];
#pragma unroll
    for (int r = 0; r < 8; r++) tm[r] = fmaxf(s0a[r], s1a[r]);
#pragma unroll
    for (int off = 1; off < 16; off <<= 1) {
#pragma unroll
      for (int r = 0; r < 8; r++) tm[r] = fmaxf(tm[r], __shfl_xor(tm[r], off, 32));
    }
    float alpha[8];
#pragma unroll
    for (int r = 0; r < 8; r++) {
      float mn = fmaxf(mrow[r], tm[r]);
      alpha[r] = __expf(mrow[r] - mn);
      mrow[r] = mn;
    }
    float p0[8], p1[8], rs[8];
#pragma unroll
    for (int r = 0; r < 8; r++) {
      p0[r] = __expf(s0a[r] - mrow[r]);
      p1[r] = __expf(s1a[r] - mrow[r]);
      rs[r] = p0[r] + p1[r];
    }
#pragma unroll
    for (int off = 1; off < 16; off <<= 1) {
#pragma unroll
      for (int r = 0; r < 8; r++) rs[r] += __shfl_xor(rs[r], off, 32);
    }
#pragma unroll
    for (int r = 0; r < 8; r++) {
      lrow[r] = lrow[r] * alpha[r] + rs[r];
      acc0[r] *= alpha[r]; acc1[r] *= alpha[r];
      acc2[r] *= alpha[r]; acc3[r] *= alpha[r];
    }
    // ---- C-layout P -> A-layout via per-wave LDS tile (in-order DS) ----
#pragma unroll
    for (int r = 0; r < 8; r++) {
      int row = r + 8 * half;
      pb[row * 32 + nl] = f2bf(p0[r]);
      pb[row * 32 + 16 + nl] = f2bf(p1[r]);
    }
    v16bf ap = loadA(pb, 32, nl, 0, half);
    // ---- P(16x32) @ V(32x64): 4 WMMAs; V stored transposed -> contiguous B ----
    const bf16_t* vb = V + (size_t)nl * Sc + kt + half * 16;
    acc0 = wmma_bf16(ap, loadBT(vb), acc0);
    acc1 = wmma_bf16(ap, loadBT(vb + (size_t)16 * Sc), acc1);
    acc2 = wmma_bf16(ap, loadBT(vb + (size_t)32 * Sc), acc2);
    acc3 = wmma_bf16(ap, loadBT(vb + (size_t)48 * Sc), acc3);
  }

  // ---- epilogue: normalize, store concat layout [b][s][h*64+dk] bf16 ----
  float inv[8];
#pragma unroll
  for (int r = 0; r < 8; r++) inv[r] = 1.0f / lrow[r];
  bf16_t* obase = obuf + (size_t)b * Sc * Dc + (size_t)h * DKc;
#pragma unroll
  for (int r = 0; r < 8; r++) {
    int srow = s0 + r + 8 * half;
    bf16_t* op = obase + (size_t)srow * Dc;
    op[nl] = f2bf(acc0[r] * inv[r]);
    op[16 + nl] = f2bf(acc1[r] * inv[r]);
    op[32 + nl] = f2bf(acc2[r] * inv[r]);
    op[48 + nl] = f2bf(acc3[r] * inv[r]);
  }
}

// ---------------- kernel 3: output projection ----------------
// grid = (B*S/16) * (768/64), block = 128
__global__ __launch_bounds__(128) void mha_oproj(
    const bf16_t* __restrict__ obuf, const bf16_t* __restrict__ wot,
    const float* __restrict__ bo, float* __restrict__ out) {
  const int lane = threadIdx.x & 31;
  const int w = threadIdx.x >> 5;
  const int half = lane >> 4;
  const int nl = lane & 15;

  int idx = blockIdx.x;
  const int NRB = (Bc * Sc) / 16;
  int rb = idx % NRB;
  int cg = idx / NRB;
  const int cb = cg * 64 + w * 16;
  const int row0 = rb * 16;
  const int m = row0 + nl;
  const int ncol = cb + nl;
  const bf16_t* brow = wot + (size_t)ncol * Dc;

  v8f acc = {};
  for (int kc = 0; kc < Dc; kc += 32) {
    v16bf a = loadA(obuf, Dc, m, kc, half);
    v16bf bm = loadBT(brow + kc + half * 16);
    acc = wmma_bf16(a, bm, acc);
  }
  const float bi = bo[ncol];
#pragma unroll
  for (int r = 0; r < 8; r++) {
    int row = row0 + r + 8 * half;
    out[(size_t)row * Dc + ncol] = acc[r] + bi;
  }
}

// ---------------- host launcher ----------------
extern "C" void kernel_launch(void* const* d_in, const int* in_sizes, int n_in,
                              void* d_out, int out_size, void* d_ws, size_t ws_size,
                              hipStream_t stream) {
  (void)in_sizes; (void)n_in; (void)out_size; (void)ws_size;
  const float* q = (const float*)d_in[0];
  const float* k = (const float*)d_in[1];
  const float* v = (const float*)d_in[2];
  const int* mask = (const int*)d_in[3];
  const float* Wq = (const float*)d_in[4];
  const float* bq = (const float*)d_in[5];
  const float* Wk = (const float*)d_in[6];
  const float* bk = (const float*)d_in[7];
  const float* Wv = (const float*)d_in[8];
  const float* bv = (const float*)d_in[9];
  const float* Wo = (const float*)d_in[10];
  const float* bo = (const float*)d_in[11];

  char* ws = (char*)d_ws;
  bf16_t* wt  = (bf16_t*)(ws + 0);          // 3*12*64*768*2  = 3,538,944
  bf16_t* wot = (bf16_t*)(ws + 3538944);    // 768*768*2      = 1,179,648
  bf16_t* xb  = (bf16_t*)(ws + 4718592);    // 3*2*2048*768*2 = 18,874,368
  bf16_t* ob  = (bf16_t*)(ws + 4718592);    // aliases xb (disjoint lifetime)
  bf16_t* qh  = (bf16_t*)(ws + 23592960);   // 24*2048*64*2   = 6,291,456
  bf16_t* kh  = (bf16_t*)(ws + 29884416);
  bf16_t* vt  = (bf16_t*)(ws + 36175872);   // total = 42,467,328 bytes

  const int BSD = Bc * Sc * Dc;
  mha_cvt<<<(3 * BSD / 8 + 255) / 256, 256, 0, stream>>>(q, k, v, xb);
  mha_prep<<<2304, 256, 0, stream>>>(Wq, Wk, Wv, Wo, wt, wot);
  mha_proj<<<3 * Hc * Bc * (Sc / 16), 128, 0, stream>>>(xb, bq, bk, bv, wt,
                                                        qh, kh, vt);
  mha_attn<<<Bc * Hc * (Sc / 64), 128, 0, stream>>>(qh, kh, vt, mask, ob);
  mha_oproj<<<(Bc * Sc / 16) * (Dc / 64), 128, 0, stream>>>(ob, wot, bo,
                                                            (float*)d_out);
}
